// Transition_52312701665879
// MI455X (gfx1250) — matrix operand order repaired and checked by reading, not compiled, not run on hardware
//
#include <hip/hip_runtime.h>

// Problem constants (from reference)
#define Bn   2
#define SRCN 8192
#define DSTN 4096
#define SDN  128
#define DDN  128

#define KTILES   (DSTN / 16)   // 256 k-tiles of 16 dst nodes
#define JSTEPS   (SRCN / 32)   // 256 j-steps of 32 src nodes
#define DTILES   9             // 8 val tiles (128 cols) + 1 state tile

typedef __attribute__((ext_vector_type(16))) __bf16 v16bf;
typedef __attribute__((ext_vector_type(8)))  float  v8f;
typedef __attribute__((ext_vector_type(4)))  unsigned int v4u;

union BF16x16 { v16bf v; v4u q[2]; };

__device__ __forceinline__ v4u ldg128(const void* p) {
  return *reinterpret_cast<const v4u*>(p);
}

__device__ __forceinline__ v8f wmma_bf16(const v16bf& a, const v16bf& b, const v8f& c) {
  // 8 args: (neg_a, A, neg_b, B, c_mod, C, reuse_a, reuse_b)
  return __builtin_amdgcn_wmma_f32_16x16x32_bf16(false, a, false, b, (short)0, c, false, false);
}

// ---------------------------------------------------------------------------
// Kernel 1a: f32 -> bf16 convert of src_val (A-operand source; rows contiguous
// in K so A tiles load directly as 2x global_load_b128 per lane).
__global__ void k_cvt_bf16(const float* __restrict__ x, __bf16* __restrict__ y, int n) {
  int i = blockIdx.x * blockDim.x + threadIdx.x;
  if (i < n) y[i] = (__bf16)x[i];
}

// Kernel 1b: pack W_route [SD,DST] into WMMA B-operand tiles.
// B (32x16, 16-bit): lane half h (lane>>4), col N = lane&15; element e<8
// holds K = h*8 + e, e>=8 holds K = 16 + h*8 + (e-8). Tile index = (kt*4+s).
__global__ void k_pack_wroute(const float* __restrict__ Wr, __bf16* __restrict__ wrB) {
  int tid = blockIdx.x * blockDim.x + threadIdx.x;       // kt*4*32 + s*32 + lane
  if (tid >= KTILES * 4 * 32) return;
  int lane = tid & 31;
  int s    = (tid >> 5) & 3;
  int kt   = tid >> 7;
  int h = lane >> 4, n = kt * 16 + (lane & 15);
  BF16x16 u;
  for (int e = 0; e < 16; e++) {
    int kk = s * 32 + ((e < 8) ? h * 8 + e : 16 + h * 8 + (e - 8)); // K index (s-dim)
    u.v[e] = (__bf16)Wr[(size_t)kk * DSTN + n];
  }
  v4u* dst = reinterpret_cast<v4u*>(wrB + (size_t)tid * 16);
  dst[0] = u.q[0]; dst[1] = u.q[1];
}

// Kernel 1c: pv = src_val @ W_val (f32).  Tiny (1 GFLOP), plain VALU.
__global__ void k_pv(const float* __restrict__ src_val, const float* __restrict__ Wv,
                     float* __restrict__ pv) {
  int i = blockIdx.x * blockDim.x + threadIdx.x;         // (b*SRC + j)*DD + d
  if (i >= Bn * SRCN * DDN) return;
  int d  = i & (DDN - 1);
  int bj = i / DDN;
  const float* row = src_val + (size_t)bj * SDN;
  float acc = 0.f;
  for (int s = 0; s < SDN; s++) acc += row[s] * Wv[(size_t)s * DDN + d];
  pv[i] = acc;
}

// Kernel 1d: pack pv (and src_state as d-tile 8, col 0) into B-operand tiles
// over (K=j 32, N=d 16).  Index = ((b*JSTEPS + js)*DTILES + dt)*32 + lane.
__global__ void k_pack_pvB(const float* __restrict__ pv, const float* __restrict__ src_state,
                           __bf16* __restrict__ pvB) {
  int tid = blockIdx.x * blockDim.x + threadIdx.x;
  if (tid >= Bn * JSTEPS * DTILES * 32) return;
  int lane = tid & 31;
  int dt   = (tid >> 5) % DTILES;
  int js   = (tid / (32 * DTILES)) % JSTEPS;
  int b    = tid / (32 * DTILES * JSTEPS);
  int h = lane >> 4, nl = lane & 15;
  BF16x16 u;
  for (int e = 0; e < 16; e++) {
    int j = js * 32 + ((e < 8) ? h * 8 + e : 16 + h * 8 + (e - 8)); // K index (j-dim)
    float v;
    if (dt < 8)      v = pv[((size_t)(b * SRCN + j)) * DDN + dt * 16 + nl];
    else             v = (nl == 0) ? src_state[b * SRCN + j] : 0.f;
    u.v[e] = (__bf16)v;
  }
  v4u* dst = reinterpret_cast<v4u*>(pvB + (size_t)tid * 16);
  dst[0] = u.q[0]; dst[1] = u.q[1];
}

// ---------------------------------------------------------------------------
// Pass 1: one wave per 16 src rows. WMMA logits over all 256 k-tiles.
// Per-lane partial exp-sums accumulate in registers (TRANS+VALU co-execute
// with WMMA); the 16-lane cross-lane reduction happens ONCE after the loop
// instead of per tile (removes the ds_bpermute storm seen in round 1).
// No max subtraction: logits ~ N(0,1) since W_route is pre-scaled by 1/sqrt(SD).
// Writes scale[b,j] = softplus(src_state)/denom.
__global__ __launch_bounds__(32) void k_denom_scale(
    const __bf16* __restrict__ sv16, const __bf16* __restrict__ wrB,
    const float* __restrict__ src_state, float* __restrict__ scale) {
  int lane = threadIdx.x;
  int jb = blockIdx.x;                                   // 0..SRC/16-1
  int b  = blockIdx.y;
  int h = lane >> 4, m = lane & 15;

  // A tiles (16 rows x 128 K), per ISA A layout: lane(h,m) row M=m,
  // elems 0..7 -> K = s*32 + h*8 + t, elems 8..15 -> K = s*32 + 16 + h*8 + t.
  const __bf16* arow = sv16 + ((size_t)(b * SRCN + jb * 16 + m)) * SDN;
  BF16x16 a[4];
  for (int s = 0; s < 4; s++) {
    int base = s * 32 + h * 8;
    a[s].q[0] = ldg128(arow + base);
    a[s].q[1] = ldg128(arow + base + 16);
  }

  float pden[8];                                         // per-lane partials
  for (int r = 0; r < 8; r++) pden[r] = 0.f;

  for (int nt = 0; nt < KTILES; nt++) {
    v8f c = {0.f, 0.f, 0.f, 0.f, 0.f, 0.f, 0.f, 0.f};
    for (int s = 0; s < 4; s++) {
      BF16x16 bb;
      const __bf16* p = wrB + ((size_t)((nt * 4 + s) * 32 + lane)) * 16;
      bb.q[0] = ldg128(p);
      bb.q[1] = ldg128(p + 8);
      c = wmma_bf16(a[s].v, bb.v, c);
    }
    // C layout: lane(h,m), vgpr r -> row M = r + 8h, col N = m.
    for (int r = 0; r < 8; r++) pden[r] += __expf(c[r]);
  }

  // single cross-lane reduction over the 16 lanes of each half
  for (int r = 0; r < 8; r++) {
    float e = pden[r];
    for (int off = 8; off > 0; off >>= 1) e += __shfl_xor(e, off, 16);
    pden[r] = e;
  }

  if (m == 0) {                                          // lanes 0 and 16 write
    for (int r = 0; r < 8; r++) {
      int j = jb * 16 + r + 8 * h;
      float st = src_state[b * SRCN + j];
      float sp = (st > 20.f) ? st : log1pf(__expf(st));  // softplus
      scale[b * SRCN + j] = sp / pden[r];
    }
  }
}

// ---------------------------------------------------------------------------
// Async stage of one j-step's 9 pvB B-tiles (9 KB) into LDS. Each lane moves
// its 32 bytes per tile with two GLOBAL_LOAD_ASYNC_TO_LDS_B128 (ISA 10.x:
// copies global memory straight into LDS, tracked by ASYNCcnt).
__device__ __forceinline__ void stage_pvB_async(const __bf16* __restrict__ pvB,
                                                int b, int js, int lane,
                                                __bf16* dstLds) {
  unsigned lbase = (unsigned)(uintptr_t)dstLds + (unsigned)lane * 32u; // LDS byte addr
  const __bf16* gbase = pvB + ((size_t)((b * JSTEPS + js) * DTILES) * 32 + lane) * 16;
  for (int dt = 0; dt < DTILES; dt++) {
    unsigned l = lbase + (unsigned)dt * 1024u;           // 32 lanes * 32B per tile
    unsigned long long g = (unsigned long long)(uintptr_t)(gbase + (size_t)dt * 512);
    asm volatile("global_load_async_to_lds_b128 %0, %1, off"
                 :: "v"(l), "v"(g) : "memory");
    asm volatile("global_load_async_to_lds_b128 %0, %1, off offset:16"
                 :: "v"(l), "v"(g) : "memory");
  }
}

// Pass 2: one wave per 16 dst rows (k-tile). Loop K=j in steps of 32:
// recompute 16x32 logits tile (WMMA), P = scale*exp, transpose via LDS into
// A-operand layout, then 9 WMMA accumulations against LDS-double-buffered
// pvB tiles (async staged one j-step ahead).
__global__ __launch_bounds__(32) void k_main(
    const __bf16* __restrict__ sv16, const __bf16* __restrict__ wrB,
    const __bf16* __restrict__ pvB, const float* __restrict__ scale,
    const float* __restrict__ dst_val, const float* __restrict__ dst_state,
    float* __restrict__ out_state, float* __restrict__ out_val) {
  __shared__ float plds[16 * 33];                        // [k_local][j_local], pad 33
  __shared__ __align__(16) __bf16 bstage[2][DTILES * 32 * 16]; // 2 x 9 KB
  int lane = threadIdx.x;
  int kt = blockIdx.x;                                   // 0..KTILES-1
  int b  = blockIdx.y;
  int h = lane >> 4, m = lane & 15;

  v8f acc[DTILES];
  for (int dt = 0; dt < DTILES; dt++)
    acc[dt] = (v8f){0.f, 0.f, 0.f, 0.f, 0.f, 0.f, 0.f, 0.f};

  // W_route B tiles for this k-tile are loop-invariant: hoist them.
  BF16x16 wb[4];
  for (int s = 0; s < 4; s++) {
    const __bf16* p = wrB + ((size_t)((kt * 4 + s) * 32 + lane)) * 16;
    wb[s].q[0] = ldg128(p);
    wb[s].q[1] = ldg128(p + 8);
  }

  stage_pvB_async(pvB, b, /*js=*/0, lane, &bstage[0][0]);
  int buf = 0;

  for (int js = 0; js < JSTEPS; js++) {
    // ---- build P tile (32 j x 16 k) in LDS ----
    for (int half = 0; half < 2; half++) {
      int j0 = js * 32 + half * 16;
      const __bf16* arow = sv16 + ((size_t)(b * SRCN + j0 + m)) * SDN;
      v8f c = {0.f, 0.f, 0.f, 0.f, 0.f, 0.f, 0.f, 0.f};
      for (int s = 0; s < 4; s++) {
        BF16x16 aa;
        int base = s * 32 + h * 8;
        aa.q[0] = ldg128(arow + base);
        aa.q[1] = ldg128(arow + base + 16);
        c = wmma_bf16(aa.v, wb[s].v, c);
      }
      const float* sp = scale + b * SRCN + j0 + 8 * h;   // rows j0+8h..+7
      for (int r = 0; r < 8; r++) {
        float p = sp[r] * __expf(c[r]);                  // logits ~O(1): safe
        // c[r] is (row j_local = half*16 + r + 8h, col k_local = m)
        plds[m * 33 + half * 16 + r + 8 * h] = p;
      }
    }
    __syncthreads();

    // ---- read back transposed into A-operand layout (M=k, K=j) ----
    BF16x16 ap;
    {
      const float* rowp = plds + m * 33;                 // row M = m
      for (int t = 0; t < 8; t++) {
        ap.v[t]     = (__bf16)rowp[h * 8 + t];           // K = h*8 + t
        ap.v[8 + t] = (__bf16)rowp[16 + h * 8 + t];      // K = 16 + h*8 + t
      }
    }
    __syncthreads();

    // ---- stage next j-step, then wait for the current buffer only ----
    if (js + 1 < JSTEPS) {
      // prefetch next A rows into L2 while we are at it (global_prefetch_b8)
      __builtin_prefetch(sv16 + ((size_t)(b * SRCN + (js + 1) * 32 + m)) * SDN, 0, 1);
      stage_pvB_async(pvB, b, js + 1, lane, &bstage[buf ^ 1][0]);
      // 18 newer loads may stay in flight; async loads complete in order
      asm volatile("s_wait_asynccnt 0x12" ::: "memory");
    } else {
      asm volatile("s_wait_asynccnt 0x0" ::: "memory");
    }

    // ---- 9 accumulating WMMAs from the staged LDS buffer ----
    for (int dt = 0; dt < DTILES; dt++) {
      BF16x16 bb;
      const v4u* p = reinterpret_cast<const v4u*>(&bstage[buf][(dt * 32 + lane) * 16]);
      bb.q[0] = p[0];
      bb.q[1] = p[1];
      acc[dt] = wmma_bf16(ap.v, bb.v, acc[dt]);
    }
    buf ^= 1;
  }

  // ---- epilogue: merge_mode='add', write concatenated outputs ----
  for (int dt = 0; dt < 8; dt++) {
    for (int r = 0; r < 8; r++) {
      int k = kt * 16 + r + 8 * h;
      int d = dt * 16 + m;
      size_t idx = ((size_t)(b * DSTN + k)) * DDN + d;
      out_val[idx] = dst_val[idx] + acc[dt][r];
    }
  }
  if (m == 0) {                                          // state column N=0
    for (int r = 0; r < 8; r++) {
      int k = kt * 16 + r + 8 * h;
      out_state[b * DSTN + k] = dst_state[b * DSTN + k] + acc[8][r];
    }
  }
}

// ---------------------------------------------------------------------------
extern "C" void kernel_launch(void* const* d_in, const int* in_sizes, int n_in,
                              void* d_out, int out_size, void* d_ws, size_t ws_size,
                              hipStream_t stream) {
  const float* src_val   = (const float*)d_in[0];
  const float* src_state = (const float*)d_in[1];
  const float* dst_val   = (const float*)d_in[2];
  const float* dst_state = (const float*)d_in[3];
  const float* W_route   = (const float*)d_in[4];
  const float* W_val     = (const float*)d_in[5];

  // outputs: new_state [B,DST] then new_val [B,DST,DD], concatenated flat
  float* out_state = (float*)d_out;
  float* out_val   = out_state + (size_t)Bn * DSTN;

  // workspace carve-up (~18.4 MB)
  char* p = (char*)d_ws;
  float*  pv    = (float*)p;  p += (size_t)Bn * SRCN * DDN * 4;
  float*  scale = (float*)p;  p += (size_t)Bn * SRCN * 4;
  __bf16* sv16  = (__bf16*)p; p += (size_t)Bn * SRCN * SDN * 2;
  __bf16* wrB   = (__bf16*)p; p += (size_t)KTILES * 4 * 32 * 16 * 2;
  __bf16* pvB   = (__bf16*)p;

  // prep
  {
    int n = Bn * SRCN * SDN;
    k_cvt_bf16<<<(n + 255) / 256, 256, 0, stream>>>(src_val, sv16, n);
  }
  {
    int n = KTILES * 4 * 32;
    k_pack_wroute<<<(n + 255) / 256, 256, 0, stream>>>(W_route, wrB);
  }
  {
    int n = Bn * SRCN * DDN;
    k_pv<<<(n + 255) / 256, 256, 0, stream>>>(src_val, W_val, pv);
  }
  {
    int n = Bn * JSTEPS * DTILES * 32;
    k_pack_pvB<<<(n + 255) / 256, 256, 0, stream>>>(pv, src_state, pvB);
  }

  // pass 1: softmax denominators -> scale = softplus(state)/denom
  k_denom_scale<<<dim3(SRCN / 16, Bn), 32, 0, stream>>>(sv16, wrB, src_state, scale);

  // pass 2: recompute logits, P^T @ [pv | state], add dst, write outputs
  k_main<<<dim3(KTILES, Bn), 32, 0, stream>>>(sv16, wrB, pvB, scale,
                                              dst_val, dst_state, out_state, out_val);
}
